// SwintransformerBlock_19430432047560
// MI455X (gfx1250) — matrix operand order repaired and checked by reading, not compiled, not run on hardware
//
#include <hip/hip_runtime.h>
#include <hip/hip_bf16.h>
#include <math.h>

// ---------------------------------------------------------------------------
// Swin shifted-window attention block, MI455X (gfx1250, wave32, WMMA).
//   B=64 H=W=56 C=128, ws=7 shift=3 heads=4 d=32  ->  4096 windows, N=49.
// One workgroup (8 wave32) per window; GEMMs on v_wmma_f32_16x16x32_bf16.
// All WMMA fragments fetched as 2 x b128 per lane (A row-major, B n-major).
// Wave index made scalar via readfirstlane so tile branches are s_cbranch.
// Per-token pixel offsets + mask labels precomputed once into LDS tables.
// ---------------------------------------------------------------------------

typedef __attribute__((ext_vector_type(16))) __bf16 v16bf;
typedef __attribute__((ext_vector_type(8)))  __bf16 v8bf;
typedef __attribute__((ext_vector_type(4)))  __bf16 v4bf;
typedef __attribute__((ext_vector_type(8)))  float  v8f;

#define WS     7
#define SHIFT  3
#define HEADS  4
#define HDIM   32
#define NTOK   49          // tokens per window
#define NPAD   64          // padded to 4 WMMA row-tiles
#define CCH    128
#define C2     256         // q|k panel width
#define C3     384
#define GRIDW  8           // 56/7 windows per axis
#define IMG    56

// LDS layout (bytes):
//   Xb    : bf16 [64][128] = 16384   (input window panel)
//   QKb   : bf16 [64][256] = 32768   (q | k, row-major [token][dim])
//   Vt    : bf16 [128][64] = 16384   (V transposed: [dim][token])
//   St    : f32  [64][64]  = 16384   (scores TRANSPOSED: [key][query])
//   Pb    : bf16 [64][64]  =  8192   (post-softmax, row-major [query][key])
//   Ob    : bf16 [64][128] = 16384
//   labrc : i32  [64]      =   256   (packed token label/r/c)
//   outoff: i32  [64]      =   256   (pixel offset of each token, elements)
#define SMEM_BYTES (16384 + 32768 + 16384 + 16384 + 8192 + 16384 + 256 + 256)

__device__ __forceinline__ v8f wmma_bf16(v16bf a, v16bf b, v8f c) {
  return __builtin_amdgcn_wmma_f32_16x16x32_bf16(
      false, a, false, b, (short)0, c, false, false);
}

// A fragment 16x32: source row-major A[m][k] at tile[m*ld + k].
// Lane layout: K = { 8*khalf + 0..7 , 16 + 8*khalf + 0..7 }  -> 2 x b128.
__device__ __forceinline__ v16bf frag_A(const __bf16* tile, int ld, int lane) {
  const __bf16* p = tile + (lane & 15) * ld + ((lane >> 4) << 3);
  v8bf lo = *(const v8bf*)p;
  v8bf hi = *(const v8bf*)(p + 16);
  return __builtin_shufflevector(lo, hi, 0,1,2,3,4,5,6,7,8,9,10,11,12,13,14,15);
}

// B fragment 32x16 from an n-major (transposed) source: B[k][n] = src[n*ld + k].
// Lane n reads K = { 8*khalf + 0..7 , 16 + 8*khalf + 0..7 } -> 2 x b128.
__device__ __forceinline__ v16bf frag_BT(const __bf16* src, int ld, int lane) {
  const __bf16* p = src + (lane & 15) * ld + ((lane >> 4) << 3);
  v8bf lo = *(const v8bf*)p;
  v8bf hi = *(const v8bf*)(p + 16);
  return __builtin_shufflevector(lo, hi, 0,1,2,3,4,5,6,7,8,9,10,11,12,13,14,15);
}

__device__ __forceinline__ int region(int u) {  // shift-mask region per axis
  return (u < IMG - WS) ? 0 : ((u < IMG - SHIFT) ? 1 : 2);
}

// ------------- prep: fp32 weights -> bf16, stored TRANSPOSED ----------------
__global__ void swin_convert_weights(const float* __restrict__ wq,
                                     const float* __restrict__ wp,
                                     __bf16* __restrict__ wqT,   // [384][128]
                                     __bf16* __restrict__ wpT) { // [128][128]
  int i = blockIdx.x * 256 + threadIdx.x;
  if (i < CCH * C3) {
    int k = i / C3, n = i % C3;
    wqT[n * CCH + k] = (__bf16)wq[i];
  }
  if (i < CCH * CCH) {
    int k = i >> 7, n = i & 127;
    wpT[n * CCH + k] = (__bf16)wp[i];
  }
}

// --------------------------- fused window kernel ----------------------------
__global__ void swin_window_kernel(const float* __restrict__ x,
                                   const float* __restrict__ bqkv,
                                   const float* __restrict__ bproj,
                                   const float* __restrict__ relt,
                                   const __bf16* __restrict__ WqT,
                                   const __bf16* __restrict__ WpT,
                                   float* __restrict__ out) {
  extern __shared__ __align__(16) char smem[];
  __bf16* Xb     = (__bf16*)smem;                  // [64][128]
  __bf16* QKb    = Xb + NPAD * CCH;                // [64][256]  q|k
  __bf16* Vt     = QKb + NPAD * C2;                // [128][64]  V^T
  float*  St     = (float*)(Vt + CCH * NPAD);      // [64][64]   S^T [key][query]
  __bf16* Pb     = (__bf16*)(St + NPAD * NPAD);    // [64][64]
  __bf16* Ob     = Pb + NPAD * NPAD;               // [64][128]
  int*    labrc  = (int*)(Ob + NPAD * CCH);        // [64]
  int*    outoff = labrc + NPAD;                   // [64]

  const int tid   = threadIdx.x;
  const int lane  = tid & 31;
  const int wave  = __builtin_amdgcn_readfirstlane(tid >> 5);  // scalar!
  const int b     = blockIdx.x;
  const int batch = b >> 6;
  const int wy    = (b >> 3) & 7;
  const int wx    = b & 7;
  const int lwy   = wy * WS, lwx = wx * WS;
  const size_t batchbase = (size_t)batch * IMG * IMG * CCH;

  // pull the L2-resident transposed bf16 weights toward this WGP
  __builtin_prefetch(WqT + (tid << 7), 0, 1);      // global_prefetch_b8
  __builtin_prefetch(WpT + (tid << 6), 0, 1);

  // ---- phase 0: per-token tables: mask label/coords + pixel offset --------
  if (tid < NPAD) {
    int lab = 0, r2 = 0, c2 = 0, off = 0;
    if (tid < NTOK) {
      r2 = tid / WS; c2 = tid % WS;
      lab = region(lwy + r2) * 3 + region(lwx + c2);
      const int gh = (lwy + r2 + SHIFT) % IMG;     // roll(x,-s); same pixel on
      const int gw = (lwx + c2 + SHIFT) % IMG;     // the way out (rolls cancel)
      off = (gh * IMG + gw) * CCH;
    }
    labrc[tid]  = lab | (r2 << 4) | (c2 << 8);
    outoff[tid] = off;
  }
  __syncthreads();

  // ---- phase 1: gather rolled window -> bf16 panel, zero-pad rows 49..63 --
  for (int idx = tid; idx < NPAD * (CCH / 4); idx += 256) {
    const int row = idx >> 5;             // 0..63
    const int c4  = (idx & 31) << 2;      // channel group of 4
    float4 f = make_float4(0.f, 0.f, 0.f, 0.f);
    if (row < NTOK)
      f = *(const float4*)(x + batchbase + outoff[row] + c4);
    v4bf v = { (__bf16)f.x, (__bf16)f.y, (__bf16)f.z, (__bf16)f.w };
    *(v4bf*)(Xb + row * CCH + c4) = v;
  }
  __syncthreads();

  // ---- phase 2: QKV GEMM [64x128]x[128x384]; q,k row-major; V -> V^T ------
  // Split into two branch-free tile loops (q|k: Nt 0..15, V: Nt 16..23);
  // both keep Mt = t&3 == wave&3 invariant under stride-8 partitioning.
  const float qscale = 0.17677669529663689f;       // 1/sqrt(32)
  {
    const int Mt = wave & 3;                       // scalar, constant per wave
    v16bf aq[4];
#pragma unroll
    for (int kk = 0; kk < 4; ++kk)
      aq[kk] = frag_A(Xb + (Mt * 16) * CCH + kk * 32, CCH, lane);

    const int rbase = Mt * 16 + ((lane >> 4) << 3);

    for (int t = wave; t < 64; t += 8) {           // q|k tiles, Nt = 0..15
      const int Nt = t >> 2;                       // scalar
      v8f acc = {};
#pragma unroll
      for (int kk = 0; kk < 4; ++kk) {
        v16bf bb = frag_BT(WqT + (size_t)(Nt * 16) * CCH + kk * 32, CCH, lane);
        acc = wmma_bf16(aq[kk], bb, acc);
      }
      const int col   = Nt * 16 + (lane & 15);
      const float bq  = bqkv[col];
      const float scl = (Nt < 8) ? qscale : 1.0f;  // scalar select: q only
#pragma unroll
      for (int r = 0; r < 8; ++r)
        QKb[(rbase + r) * C2 + col] = (__bf16)((acc[r] + bq) * scl);
    }

    for (int t = wave; t < 32; t += 8) {           // V tiles, Nt = 16..23
      const int Nt = 16 + (t >> 2);                // scalar
      v8f acc = {};
#pragma unroll
      for (int kk = 0; kk < 4; ++kk) {
        v16bf bb = frag_BT(WqT + (size_t)(Nt * 16) * CCH + kk * 32, CCH, lane);
        acc = wmma_bf16(aq[kk], bb, acc);
      }
      const int col  = Nt * 16 + (lane & 15);
      const float bq = bqkv[col];
      v8bf pk;
#pragma unroll
      for (int r = 0; r < 8; ++r) pk[r] = (__bf16)(acc[r] + bq);
      *(v8bf*)(Vt + (col - C2) * NPAD + rbase) = pk;   // V^T, b128 store
    }
  }
  __syncthreads();

  // ---- phase 3: per-head attention ----------------------------------------
#pragma unroll 1
  for (int h = 0; h < HEADS; ++h) {
    // S = q_h @ k_h^T : K = d = 32 -> single WMMA per tile; store S^T packed
    {
      const int Mt = wave & 3;                     // scalar
      v16bf aS = frag_A(QKb + (Mt * 16) * C2 + h * HDIM, C2, lane);
      for (int t = wave; t < 16; t += 8) {
        const int Nt = t >> 2;                     // scalar
        v16bf bb = frag_BT(QKb + (Nt * 16) * C2 + CCH + h * HDIM, C2, lane);
        v8f acc = {};
        acc = wmma_bf16(aS, bb, acc);
        const int col   = Nt * 16 + (lane & 15);   // key index
        const int rbase = Mt * 16 + ((lane >> 4) << 3);
        const bool ok   = col < NTOK;              // pad keys -> -inf
        float4 lo = make_float4(ok ? acc[0] : -1e30f, ok ? acc[1] : -1e30f,
                                ok ? acc[2] : -1e30f, ok ? acc[3] : -1e30f);
        float4 hi = make_float4(ok ? acc[4] : -1e30f, ok ? acc[5] : -1e30f,
                                ok ? acc[6] : -1e30f, ok ? acc[7] : -1e30f);
        float* dst = St + col * NPAD + rbase;      // contiguous rows per lane
        *(float4*)dst       = lo;
        *(float4*)(dst + 4) = hi;
      }
    }
    __syncthreads();

    // row softmax (bank-conflict-free: thread `row` reads St[m*64+row]),
    // then (faithful to source) + rel-bias + shift-mask -> bf16, v8 stores
    if (tid < NPAD) {
      const int row = tid;
      if (row < NTOK) {
        float mx = -1e30f;
#pragma unroll 4
        for (int m = 0; m < NTOK; ++m) mx = fmaxf(mx, St[m * NPAD + row]);
        float sum = 0.f;
#pragma unroll 4
        for (int m = 0; m < NTOK; ++m) {
          float e = __expf(St[m * NPAD + row] - mx);
          St[m * NPAD + row] = e;                  // cache exp in place
          sum += e;
        }
        const float inv = 1.0f / sum;
        const int pk1 = labrc[row];
        const int lab1 = pk1 & 15;
        const int r1 = (pk1 >> 4) & 15, c1 = (pk1 >> 8) & 15;
        const float* relh = relt + h;
#pragma unroll 1
        for (int mb = 0; mb < NPAD; mb += 8) {
          v8bf pk;
#pragma unroll
          for (int j = 0; j < 8; ++j) {
            const int m = mb + j;
            float p = 0.f;
            if (m < NTOK) {
              p = St[m * NPAD + row] * inv;
              const int pk2 = labrc[m];
              const int r2 = (pk2 >> 4) & 15, c2 = (pk2 >> 8) & 15;
              p += relh[((r1 - r2 + WS - 1) * (2 * WS - 1) + (c1 - c2 + WS - 1)) * HEADS];
              if ((pk2 & 15) != lab1) p -= 100.0f;
            }
            pk[j] = (__bf16)p;
          }
          *(v8bf*)(Pb + row * NPAD + mb) = pk;
        }
      } else {
        v8bf z = {};
#pragma unroll 1
        for (int mb = 0; mb < NPAD; mb += 8) *(v8bf*)(Pb + row * NPAD + mb) = z;
      }
    }
    __syncthreads();

    // O_h = P @ V_h : [64x64]x[64x32], B frags from V^T (contiguous)
    {
      const int Nt = wave & 1, Mt = wave >> 1;     // scalar; one tile per wave
      v8f acc = {};
#pragma unroll
      for (int kk = 0; kk < 4; ++kk) {
        v16bf a  = frag_A (Pb + (Mt * 16) * NPAD + kk * 32, NPAD, lane);
        v16bf bb = frag_BT(Vt + (h * HDIM + Nt * 16) * NPAD + kk * 32, NPAD, lane);
        acc = wmma_bf16(a, bb, acc);
      }
      const int col   = h * HDIM + Nt * 16 + (lane & 15);
      const int rbase = Mt * 16 + ((lane >> 4) << 3);
#pragma unroll
      for (int r = 0; r < 8; ++r)
        Ob[(rbase + r) * CCH + col] = (__bf16)acc[r];
    }
    __syncthreads();
  }

  // ---- phase 4: proj GEMM + bias, scatter via precomputed pixel offsets ---
  {
    const int Mt = wave & 3;                       // scalar
    v16bf aO[4];
#pragma unroll
    for (int kk = 0; kk < 4; ++kk)
      aO[kk] = frag_A(Ob + (Mt * 16) * CCH + kk * 32, CCH, lane);

    for (int t = wave; t < 32; t += 8) {
      const int Nt = t >> 2;                       // scalar
      v8f acc = {};
#pragma unroll
      for (int kk = 0; kk < 4; ++kk) {
        v16bf bb = frag_BT(WpT + (Nt * 16) * CCH + kk * 32, CCH, lane);
        acc = wmma_bf16(aO[kk], bb, acc);
      }
      const int col   = Nt * 16 + (lane & 15);
      const float bp  = bproj[col];
      const int rbase = Mt * 16 + ((lane >> 4) << 3);
#pragma unroll
      for (int r = 0; r < 8; ++r) {
        const int row = rbase + r;
        if (row < NTOK)
          out[batchbase + outoff[row] + col] = acc[r] + bp;
      }
    }
  }
}

// ---------------------------------------------------------------------------
extern "C" void kernel_launch(void* const* d_in, const int* in_sizes, int n_in,
                              void* d_out, int out_size, void* d_ws, size_t ws_size,
                              hipStream_t stream) {
  const float* x      = (const float*)d_in[0];
  const float* w_qkv  = (const float*)d_in[1];
  const float* b_qkv  = (const float*)d_in[2];
  const float* w_proj = (const float*)d_in[3];
  const float* b_proj = (const float*)d_in[4];
  const float* relt   = (const float*)d_in[5];
  float* out          = (float*)d_out;

  __bf16* WqT = (__bf16*)d_ws;                              // [384][128] bf16
  __bf16* WpT = (__bf16*)((char*)d_ws + (size_t)CCH * C3 * sizeof(__bf16));

  swin_convert_weights<<<(CCH * C3 + 255) / 256, 256, 0, stream>>>(
      w_qkv, w_proj, WqT, WpT);

  const int n_windows = 64 * GRIDW * GRIDW;                 // 4096
  swin_window_kernel<<<n_windows, 256, SMEM_BYTES, stream>>>(
      x, b_qkv, b_proj, relt, WqT, WpT, out);
}
